// AdaptiveFunctionBlock_67456756351400
// MI455X (gfx1250) — matrix operand order repaired and checked by reading, not compiled, not run on hardware
//
#include <hip/hip_runtime.h>
#include <hip/hip_bf16.h>

typedef __attribute__((ext_vector_type(16))) __bf16 v16bf;
typedef __attribute__((ext_vector_type(8)))  float  v8f;
typedef unsigned int u32x4 __attribute__((ext_vector_type(4)));
typedef int          i32x4 __attribute__((ext_vector_type(4)));
typedef int          i32x8 __attribute__((ext_vector_type(8)));

#define NN 8192
#define DD 256
#define DA 64

#if defined(__HIP_DEVICE_COMPILE__) && defined(__has_builtin)
#if __has_builtin(__builtin_amdgcn_tensor_load_to_lds)
#define HAVE_TDM 1
#else
#define HAVE_TDM 0
#endif
#else
#define HAVE_TDM 0
#endif

static __device__ __forceinline__ __bf16 f2b(float f) { return (__bf16)f; }

// ---------------- prior: R = (S + I) / rowsum, emitted in bf16 ----------------
__global__ __launch_bounds__(256) void k_rownorm(const float* __restrict__ S,
                                                 __bf16* __restrict__ R16) {
  __shared__ float red[256];
  const int row = blockIdx.x;
  const int t = threadIdx.x;
  const float* srow = S + (size_t)row * NN;
  float v[32];
  float s = 0.f;
#pragma unroll
  for (int i = 0; i < 32; ++i) {
    int c = i * 256 + t;
    float x = srow[c];
    if (c == row) x += 1.f;
    v[i] = x; s += x;
  }
  red[t] = s; __syncthreads();
  for (int o = 128; o > 0; o >>= 1) { if (t < o) red[t] += red[t + o]; __syncthreads(); }
  const float inv = 1.f / red[0];
  __bf16* rrow = R16 + (size_t)row * NN;
#pragma unroll
  for (int i = 0; i < 32; ++i) rrow[i * 256 + t] = f2b(v[i] * inv);
}

// ---------------- fold Wq = W2 @ W1 (64x256), store bf16 ----------------
__global__ __launch_bounds__(256) void k_wq(const float* __restrict__ W1,
                                            const float* __restrict__ W2,
                                            __bf16* __restrict__ Wq16) {
  const int o = blockIdx.x, i = threadIdx.x;
  float s = 0.f;
#pragma unroll
  for (int m = 0; m < 64; ++m) s = fmaf(W2[o * 64 + m], W1[m * 256 + i], s);
  Wq16[o * 256 + i] = f2b(s);
}

// ---------------- generic f32 -> bf16 ----------------
__global__ void k_cvt(const float* __restrict__ src, __bf16* __restrict__ dst, int n) {
  int i = blockIdx.x * 256 + threadIdx.x;
  if (i < n) dst[i] = f2b(src[i]);
}

// ---------------- X (N x 256) f32 -> bf16 row-major + transposed ----------------
__global__ __launch_bounds__(256) void k_cvt_x(const float* __restrict__ src,
                                               __bf16* __restrict__ d,
                                               __bf16* __restrict__ dT) {
  int i = blockIdx.x * 256 + threadIdx.x;
  int r = i >> 8, c = i & 255;
  __bf16 v = f2b(src[i]);
  d[i] = v;
  dT[(size_t)c * NN + r] = v;
}

// ---------------- WMMA bf16 GEMM: C[M x N'] = A[M x K] * B([N'][K] col-contig) ----
// Block tile 128x64, K-chunk 32, 256 threads = 8 waves (4 M x 2 N), each wave 32x32.
// Double-buffered LDS; A tile staged by the Tensor Data Mover (TENSORcnt), B tile by
// GLOBAL_LOAD_ASYNC_TO_LDS_B128 (ASYNCcnt); DMA for chunk c+1 overlaps WMMA on chunk c.
__global__ __launch_bounds__(256) void k_gemm(const __bf16* __restrict__ A,
                                              const __bf16* __restrict__ B,
                                              float* __restrict__ C,
                                              int K, int accumulate) {
  __shared__ __align__(16) __bf16 At[2][128][32];
  __shared__ __align__(16) __bf16 Bt[2][64][32];
  const int t = threadIdx.x;
  const int lane = t & 31;
  const int w = t >> 5;
  const int waveM = (w & 3) * 32;
  const int waveN = (w >> 2) * 32;
  const int rowBase = blockIdx.y * 128;
  const int colBase = blockIdx.x * 64;
  const int ldc = (int)gridDim.x * 64;

  const __bf16* Ab = A + (size_t)rowBase * K;
  const __bf16* Bb = B + (size_t)colBase * K;

  const v8f zero = {0.f, 0.f, 0.f, 0.f, 0.f, 0.f, 0.f, 0.f};
  v8f acc00 = zero, acc01 = zero, acc10 = zero, acc11 = zero;

  // staging: 16B (8 x bf16) chunks. B: 256 chunks (1/thread). A: TDM tile (or 2/thread async)
  const int ar0 = t >> 2,          ao0 = (t & 3) * 8;
  const int ar1 = (t + 256) >> 2,  ao1 = ((t + 256) & 3) * 8;
  const int brc = t >> 2,          boc = (t & 3) * 8;

  // wave-relative LDS byte offsets (low 32 bits of the generic LDS pointer)
  const unsigned lB[2]  = {(unsigned)(uintptr_t)&Bt[0][brc][boc],
                           (unsigned)(uintptr_t)&Bt[1][brc][boc]};
#if HAVE_TDM
  const unsigned ldsAbase[2] = {(unsigned)(uintptr_t)&At[0][0][0],
                                (unsigned)(uintptr_t)&At[1][0][0]};
  // D# group1 (ISA 8.4): data_size=2B; tensor_dim0=32, tensor_dim1=128 (== tile dims,
  // no OOB); tile_dim0=32, tile_dim1=128, tile_dim2=0; tensor_dim0_stride=K elements.
  i32x8 g1d;
  g1d[0] = 1 << 16;          // data_size=1 (2 bytes); mask/barrier/iterate/pad = 0
  g1d[1] = 32 << 16;         // tensor_dim0[15:0]
  g1d[2] = 128 << 16;        // tensor_dim1[15:0]
  g1d[3] = 32 << 16;         // tile_dim0
  g1d[4] = 128;              // tile_dim1 (tile_dim2 = 0)
  g1d[5] = K;                // tensor_dim0_stride[31:0]
  g1d[6] = 0;
  g1d[7] = 0;
  const i32x4 z4 = {0, 0, 0, 0};
  const i32x8 z8 = {0, 0, 0, 0, 0, 0, 0, 0};
  (void)z8;
#else
  const unsigned lA0[2] = {(unsigned)(uintptr_t)&At[0][ar0][ao0],
                           (unsigned)(uintptr_t)&At[1][ar0][ao0]};
  const unsigned lA1[2] = {(unsigned)(uintptr_t)&At[0][ar1][ao1],
                           (unsigned)(uintptr_t)&At[1][ar1][ao1]};
#endif

  auto stage = [&](int buf, int k0) {
#if HAVE_TDM
    if (w == 0) {  // one TDM descriptor moves the whole 128x32 A tile
      unsigned long long ga = (unsigned long long)(uintptr_t)Ab + (unsigned long long)(unsigned)k0 * 2ull;
      u32x4 g0;
      g0[0] = 1u;                                            // count=1, user mode
      g0[1] = ldsAbase[buf];                                 // lds_addr
      g0[2] = (unsigned)ga;                                  // global_addr[31:0]
      g0[3] = ((unsigned)(ga >> 32) & 0x01FFFFFFu) | 0x80000000u;  // ga[56:32] | type=2
#if __clang_major__ >= 23
      __builtin_amdgcn_tensor_load_to_lds(g0, g1d, z4, z4, z8, 0);
#else
      __builtin_amdgcn_tensor_load_to_lds(g0, g1d, z4, z4, 0);
#endif
    }
#else
    asm volatile("global_load_async_to_lds_b128 %0, %1, off"
                 :: "v"(lA0[buf]), "v"(&Ab[(size_t)ar0 * K + k0 + ao0]) : "memory");
    asm volatile("global_load_async_to_lds_b128 %0, %1, off"
                 :: "v"(lA1[buf]), "v"(&Ab[(size_t)ar1 * K + k0 + ao1]) : "memory");
#endif
    asm volatile("global_load_async_to_lds_b128 %0, %1, off"
                 :: "v"(lB[buf]), "v"(&Bb[(size_t)brc * K + k0 + boc]) : "memory");
  };

  const int nc = K >> 5;
  stage(0, 0);
  for (int c = 0; c < nc; ++c) {
    const int cur = c & 1;
    const bool more = (c + 1) < nc;
    if (more) stage(cur ^ 1, (c + 1) << 5);  // DMA next tile while computing this one
    // drain this tile's group only: per wave, <= newer_outstanding may remain in flight
    if (more) {
#if HAVE_TDM
      asm volatile("s_wait_asynccnt 0x1" ::: "memory");
      __builtin_amdgcn_s_wait_tensorcnt(1);
#else
      asm volatile("s_wait_asynccnt 0x3" ::: "memory");
#endif
    } else {
      asm volatile("s_wait_asynccnt 0x0" ::: "memory");
#if HAVE_TDM
      __builtin_amdgcn_s_wait_tensorcnt(0);
#endif
    }
    __syncthreads();  // tile visible to all waves

    // ISA 7.12.2 fragment layouts (wave32):
    //  A 16x32 bf16: lanes 0-15 row=lane, K {0-7,16-23}; lanes 16-31 row=lane-16, K {8-15,24-31}
    //  B 32x16 bf16: lanes 0-15 col=lane, K {0-7,16-23}; lanes 16-31 col=lane-16, K {8-15,24-31}
    const int mrow = lane & 15;
    const int akb  = (lane >> 4) * 8;
    const int bkb  = (lane >> 4) * 8;
    v16bf a0, a1, b0, b1;
    ((uint4*)&a0)[0] = *(const uint4*)&At[cur][waveM + mrow][akb];
    ((uint4*)&a0)[1] = *(const uint4*)&At[cur][waveM + mrow][akb + 16];
    ((uint4*)&a1)[0] = *(const uint4*)&At[cur][waveM + 16 + mrow][akb];
    ((uint4*)&a1)[1] = *(const uint4*)&At[cur][waveM + 16 + mrow][akb + 16];
    ((uint4*)&b0)[0] = *(const uint4*)&Bt[cur][waveN + mrow][bkb];
    ((uint4*)&b0)[1] = *(const uint4*)&Bt[cur][waveN + mrow][bkb + 16];
    ((uint4*)&b1)[0] = *(const uint4*)&Bt[cur][waveN + 16 + mrow][bkb];
    ((uint4*)&b1)[1] = *(const uint4*)&Bt[cur][waveN + 16 + mrow][bkb + 16];

    acc00 = __builtin_amdgcn_wmma_f32_16x16x32_bf16(false, a0, false, b0, (short)0, acc00, false, false);
    acc01 = __builtin_amdgcn_wmma_f32_16x16x32_bf16(false, a0, false, b1, (short)0, acc01, false, false);
    acc10 = __builtin_amdgcn_wmma_f32_16x16x32_bf16(false, a1, false, b0, (short)0, acc10, false, false);
    acc11 = __builtin_amdgcn_wmma_f32_16x16x32_bf16(false, a1, false, b1, (short)0, acc11, false, false);

    __syncthreads();  // release this buffer for the DMA two chunks ahead
  }

  // C/D layout: VGPR r -> lanes 0-15: M=r, N=lane; lanes 16-31: M=r+8, N=lane-16
  const int cn = lane & 15;
  const int mo = (lane >> 4) * 8;
#pragma unroll
  for (int r = 0; r < 8; ++r) {
    const int m0 = rowBase + waveM + r + mo;
    const int m1 = m0 + 16;
    const size_t i00 = (size_t)m0 * ldc + colBase + waveN + cn;
    const size_t i01 = i00 + 16;
    const size_t i10 = (size_t)m1 * ldc + colBase + waveN + cn;
    const size_t i11 = i10 + 16;
    float v00 = acc00[r], v01 = acc01[r], v10 = acc10[r], v11 = acc11[r];
    if (accumulate) { v00 += C[i00]; v01 += C[i01]; v10 += C[i10]; v11 += C[i11]; }
    C[i00] = v00; C[i01] = v01; C[i10] = v10; C[i11] = v11;
  }
}

// ---------------- fused logits + softmax + nucleus(top-p) per row -> A16 bf16 ----
__global__ __launch_bounds__(1024) void k_attn(const float* __restrict__ q,
                                               const float* __restrict__ kmat,
                                               __bf16* __restrict__ A16) {
  __shared__ float sh_s[NN];       // 32 KB: logits -> probs -> sorted
  __shared__ float sh_part[1024];  // reductions / scan partials
  __shared__ float sh_q[64];
  __shared__ float sh_out[2];
  const int row = blockIdx.x;
  const int t = threadIdx.x;
  if (t < 64) sh_q[t] = q[row * 64 + t];
  __syncthreads();

  float l_loc[8];
  float lmax = -3.0e38f;
#pragma unroll
  for (int i = 0; i < 8; ++i) {
    const int j = i * 1024 + t;
    const float* kj = kmat + (size_t)j * 64;
    float s = 0.f;
#pragma unroll
    for (int u = 0; u < 64; ++u) s = fmaf(sh_q[u], kj[u], s);
    l_loc[i] = s;  // TAU == 1
    lmax = fmaxf(lmax, s);
  }
  sh_part[t] = lmax; __syncthreads();
  for (int o = 512; o > 0; o >>= 1) { if (t < o) sh_part[t] = fmaxf(sh_part[t], sh_part[t + o]); __syncthreads(); }
  const float m = sh_part[0]; __syncthreads();

  float zs = 0.f;
#pragma unroll
  for (int i = 0; i < 8; ++i) { l_loc[i] = __expf(l_loc[i] - m); zs += l_loc[i]; }
  sh_part[t] = zs; __syncthreads();
  for (int o = 512; o > 0; o >>= 1) { if (t < o) sh_part[t] += sh_part[t + o]; __syncthreads(); }
  const float invZ = 1.f / sh_part[0]; __syncthreads();

  float p_loc[8];
#pragma unroll
  for (int i = 0; i < 8; ++i) {
    const float p = (float)f2b(l_loc[i] * invZ);  // bf16-round once, consistently
    p_loc[i] = p;
    sh_s[i * 1024 + t] = p;
  }
  __syncthreads();

  // bitonic sort, descending, 8192 elements in LDS
  for (unsigned size = 2; size <= NN; size <<= 1) {
    for (unsigned stride = size >> 1; stride > 0; stride >>= 1) {
#pragma unroll 1
      for (unsigned p0 = (unsigned)t; p0 < NN / 2; p0 += 1024) {
        const unsigned lo = ((p0 & ~(stride - 1)) << 1) | (p0 & (stride - 1));
        const unsigned hi = lo + stride;
        const float a = sh_s[lo], b = sh_s[hi];
        const bool desc = ((lo & size) == 0);
        if (desc ? (a < b) : (a > b)) { sh_s[lo] = b; sh_s[hi] = a; }
      }
      __syncthreads();
    }
  }

  // inclusive scan of sorted probs (8/thread local + Hillis-Steele across 1024)
  float loc[8]; float run = 0.f;
#pragma unroll
  for (int i = 0; i < 8; ++i) { run += sh_s[t * 8 + i]; loc[i] = run; }
  sh_part[t] = run; __syncthreads();
  for (int o = 1; o < 1024; o <<= 1) {
    const float v = (t >= o) ? sh_part[t - o] : 0.f;
    __syncthreads();
    sh_part[t] += v;
    __syncthreads();
  }
  const float base = sh_part[t] - run;
  if (t == 0) { sh_out[0] = 0.f; sh_out[1] = 1.f; }
  __syncthreads();
#pragma unroll
  for (int i = 0; i < 8; ++i) {
    const int idx = t * 8 + i;
    const float val = sh_s[idx];
    const float incl = base + loc[i];
    const float excl = incl - val;
    // keep while exclusive prefix < p; boundary element records threshold + kept mass
    if (excl < 0.9f && (idx == NN - 1 || incl >= 0.9f)) { sh_out[0] = val; sh_out[1] = incl; }
  }
  __syncthreads();
  const float thr = sh_out[0];
  const float Ms = fmaxf(sh_out[1], 1e-12f);
  __bf16* arow = A16 + (size_t)row * NN;
#pragma unroll
  for (int i = 0; i < 8; ++i) {
    const int j = i * 1024 + t;
    const float p = p_loc[i];
    arow[j] = (p >= thr) ? f2b(p / Ms) : f2b(0.f);
  }
}

// ---------------- h = X + Z; LayerNorm ----------------
__global__ __launch_bounds__(256) void k_ln(const float* __restrict__ X,
                                            const float* __restrict__ Z,
                                            const float* __restrict__ g,
                                            const float* __restrict__ b,
                                            float* __restrict__ out) {
  __shared__ float red[256];
  const int row = blockIdx.x, t = threadIdx.x;
  const size_t i = (size_t)row * 256 + t;
  const float h = X[i] + Z[i];
  red[t] = h; __syncthreads();
  for (int o = 128; o > 0; o >>= 1) { if (t < o) red[t] += red[t + o]; __syncthreads(); }
  const float mu = red[0] * (1.f / 256.f); __syncthreads();
  const float d = h - mu;
  red[t] = d * d; __syncthreads();
  for (int o = 128; o > 0; o >>= 1) { if (t < o) red[t] += red[t + o]; __syncthreads(); }
  const float var = red[0] * (1.f / 256.f);
  out[i] = d * rsqrtf(var + 1e-5f) * g[t] + b[t];
}

extern "C" void kernel_launch(void* const* d_in, const int* in_sizes, int n_in,
                              void* d_out, int out_size, void* d_ws, size_t ws_size,
                              hipStream_t stream) {
  const float* X  = (const float*)d_in[0];
  const float* S  = (const float*)d_in[1];
  const float* W1 = (const float*)d_in[2];
  const float* W2 = (const float*)d_in[3];
  const float* W3 = (const float*)d_in[4];
  const float* U1 = (const float*)d_in[5];
  const float* U2 = (const float*)d_in[6];
  const float* lg = (const float*)d_in[7];
  const float* lb = (const float*)d_in[8];
  float* out = (float*)d_out;

  char* w = (char*)d_ws;
  size_t off = 0;
  auto alloc = [&](size_t bytes) -> void* {
    void* p = w + off;
    off += (bytes + 255) & ~(size_t)255;
    return p;
  };
  const size_t NNb16 = (size_t)NN * NN * 2;   // 128 MB
  const size_t NXf32 = (size_t)NN * DD * 4;   // 8 MB
  const size_t NXb16 = (size_t)NN * DD * 2;   // 4 MB

  __bf16* R16   = (__bf16*)alloc(NNb16);
  __bf16* A16   = (__bf16*)alloc(NNb16);
  float*  XfA   = (float*)alloc(NXf32);
  float*  XfB   = (float*)alloc(NXf32);
  float*  XaA   = (float*)alloc(NXf32);
  float*  XaB   = (float*)alloc(NXf32);
  float*  Z     = (float*)alloc(NXf32);
  __bf16* Xf16  = (__bf16*)alloc(NXb16);
  __bf16* Xf16T = (__bf16*)alloc(NXb16);
  __bf16* Xa16  = (__bf16*)alloc(NXb16);
  __bf16* Xa16T = (__bf16*)alloc(NXb16);
  float*  qb    = (float*)alloc((size_t)NN * DA * 4);
  float*  kb    = (float*)alloc((size_t)NN * DA * 4);
  __bf16* Wq16  = (__bf16*)alloc((size_t)DA * DD * 2);
  __bf16* W316  = (__bf16*)alloc((size_t)DA * DD * 2);
  __bf16* U116  = (__bf16*)alloc((size_t)2 * DD * DD * 2);
  __bf16* U216  = (__bf16*)alloc((size_t)2 * DD * DD * 2);
  if (off > ws_size) return;  // workspace too small; deterministic no-op

  // prep
  hipMemcpyAsync(XfA, X, NXf32, hipMemcpyDeviceToDevice, stream);
  hipMemcpyAsync(XaA, X, NXf32, hipMemcpyDeviceToDevice, stream);
  k_rownorm<<<NN, 256, 0, stream>>>(S, R16);
  k_wq<<<64, 256, 0, stream>>>(W1, W2, Wq16);
  k_cvt<<<(DA * DD + 255) / 256, 256, 0, stream>>>(W3, W316, DA * DD);
  k_cvt<<<(2 * DD * DD + 255) / 256, 256, 0, stream>>>(U1, U116, 2 * DD * DD);
  k_cvt<<<(2 * DD * DD + 255) / 256, 256, 0, stream>>>(U2, U216, 2 * DD * DD);

  float *Xf = XfA, *XfN = XfB, *Xa = XaA, *XaN = XaB;
  for (int n = 0; n < 2; ++n) {
    k_cvt_x<<<NN, 256, 0, stream>>>(Xa, Xa16, Xa16T);
    k_cvt_x<<<NN, 256, 0, stream>>>(Xf, Xf16, Xf16T);
    // q = Xa @ (W2 W1)^T ; k = Xa @ W3^T   (N'=64)
    k_gemm<<<dim3(1, NN / 128), 256, 0, stream>>>(Xa16, Wq16, qb, DD, 0);
    k_gemm<<<dim3(1, NN / 128), 256, 0, stream>>>(Xa16, W316, kb, DD, 0);
    // nucleus attention adjacency (bf16 row-stochastic-after-mask)
    k_attn<<<NN, 1024, 0, stream>>>(qb, kb, A16);
    // diffusion: Xf' = R @ Xf ; Xa' = A @ Xa   (K = 8192)
    k_gemm<<<dim3(DD / 64, NN / 128), 256, 0, stream>>>(R16, Xf16T, XfN, NN, 0);
    k_gemm<<<dim3(DD / 64, NN / 128), 256, 0, stream>>>(A16, Xa16T, XaN, NN, 0);
    // refresh bf16 copies of the new states
    k_cvt_x<<<NN, 256, 0, stream>>>(XfN, Xf16, Xf16T);
    k_cvt_x<<<NN, 256, 0, stream>>>(XaN, Xa16, Xa16T);
    // Z (+)= Xf' @ U1[n]^T + Xa' @ U2[n]^T
    k_gemm<<<dim3(DD / 64, NN / 128), 256, 0, stream>>>(Xf16, U116 + (size_t)n * DD * DD, Z, DD, n == 0 ? 0 : 1);
    k_gemm<<<dim3(DD / 64, NN / 128), 256, 0, stream>>>(Xa16, U216 + (size_t)n * DD * DD, Z, DD, 1);
    float* tmp;
    tmp = Xf; Xf = XfN; XfN = tmp;
    tmp = Xa; Xa = XaN; XaN = tmp;
  }
  k_ln<<<NN, 256, 0, stream>>>(X, Z, lg, lb, out);
}
